// MoE_25984552141451
// MI455X (gfx1250) — matrix operand order repaired and checked by reading, not compiled, not run on hardware
//
#include <hip/hip_runtime.h>

// ---------------------------------------------------------------------------
// MoE (E=16, top-2, group-limited) for MI455X / gfx1250.
// bf16 WMMA 16x16x32, pre-converted+transposed bf16 weights, async LDS copies
// (ASYNCcnt double-buffered), expert-compacted tiles, atomic f32 combine.
// ---------------------------------------------------------------------------

#define T_TOK   4096
#define DIM_    1024
#define NE      16
#define INTER_  512
#define NSHI    1024
#define MAX_TILES 528
#define KT      64            // K-step (two WMMA K-slices)
#define LDT     72            // LDS row stride in halfs (16B-friendly, skewed)

typedef __attribute__((ext_vector_type(16))) __bf16       v16bf;
typedef __attribute__((ext_vector_type(8)))  float        v8f;
typedef __attribute__((ext_vector_type(4)))  unsigned int v4u;
typedef int gv4i __attribute__((vector_size(16)));        // matches builtin V4i

#define WMMA_BF16(A, B, C) \
  __builtin_amdgcn_wmma_f32_16x16x32_bf16(false, (A), false, (B), (short)0, (C), false, false)

// ------------------------- async copy plumbing ------------------------------
#if defined(__has_builtin)
#  if __has_builtin(__builtin_amdgcn_global_load_async_to_lds_b128)
#    define USE_ASYNC 1
#  else
#    define USE_ASYNC 0
#  endif
#else
#  define USE_ASYNC 0
#endif

__device__ __forceinline__ void cp16(unsigned short* ldst, const unsigned short* gsrc) {
#if USE_ASYNC
  __builtin_amdgcn_global_load_async_to_lds_b128(
      (__attribute__((address_space(1))) gv4i*)(unsigned long long)gsrc,
      (__attribute__((address_space(3))) gv4i*)(unsigned int)(unsigned long long)ldst,
      0, 0);
#else
  *(uint4*)ldst = *(const uint4*)gsrc;
#endif
}

template <int N>
__device__ __forceinline__ void waitAsync() {
#if USE_ASYNC
#  if __has_builtin(__builtin_amdgcn_s_wait_asynccnt)
  __builtin_amdgcn_s_wait_asynccnt(N);
#  else
  asm volatile("s_wait_asynccnt %0" :: "i"(N) : "memory");
#  endif
#endif
}

// ------------------------------ bf16 utils ----------------------------------
__device__ __forceinline__ unsigned short f2bf(float f) {
  unsigned int u = __float_as_uint(f);
  return (unsigned short)((u + 0x7FFFu + ((u >> 16) & 1u)) >> 16);
}
__device__ __forceinline__ unsigned int pack2bf(float a, float b) {
  return (unsigned int)f2bf(a) | ((unsigned int)f2bf(b) << 16);
}

// A fragment: 16x32 bf16. lane L: M=L%16, h=L/16 -> K in [8h,8h+8) u [16+8h,+8)
__device__ __forceinline__ v16bf fragA(const unsigned short* As, int lane, int kh) {
  int m = lane & 15, h = lane >> 4;
  const unsigned short* p = As + m * LDT + kh * 32 + 8 * h;
  union { v4u u[2]; v16bf f; } u;
  u.u[0] = *(const v4u*)(p);
  u.u[1] = *(const v4u*)(p + 16);
  return u.f;
}
// B fragment: 32x16 bf16, LDS [n][k]: lane L: N=nsub+L%16, K in [8h,8h+8) u [16+8h,+8)
__device__ __forceinline__ v16bf fragB(const unsigned short* Bs, int nsub, int lane, int kh) {
  int n = nsub + (lane & 15), h = lane >> 4;
  const unsigned short* p = Bs + n * LDT + kh * 32 + 8 * h;
  union { v4u u[2]; v16bf f; } u;
  u.u[0] = *(const v4u*)(p);
  u.u[1] = *(const v4u*)(p + 16);
  return u.f;
}

// stage a 64(n) x 64(k) bf16 panel, k-contiguous rows, 4 x b128 per thread
__device__ __forceinline__ void stageB64(unsigned short* Bs, const unsigned short* Wb,
                                         int ldk, int k0, int tid) {
#pragma unroll
  for (int i = 0; i < 4; ++i) {
    int lin = i * 128 + tid;
    int nn = lin >> 3, kc = (lin & 7) << 3;
    cp16(Bs + nn * LDT + kc, Wb + (size_t)nn * ldk + k0 + kc);
  }
}

// ------------------------------ gate ---------------------------------------
__global__ __launch_bounds__(256) void gate_kernel(
    const float* __restrict__ x, const float* __restrict__ gw,
    const float* __restrict__ gb, int* __restrict__ idx_out,
    float* __restrict__ w_out, int* __restrict__ counts) {
  __shared__ float gws[NE * DIM_];
  int tid = threadIdx.x;
  const float4* gw4 = (const float4*)gw;
  float4* gws4 = (float4*)gws;
  for (int i = tid; i < NE * 256; i += 256) gws4[i] = gw4[i];
  __syncthreads();

  int wave = tid >> 5, lane = tid & 31;
  int t = blockIdx.x * 8 + wave;

  float acc[NE];
#pragma unroll
  for (int e = 0; e < NE; ++e) acc[e] = 0.f;
  const float4* xr = (const float4*)(x + (size_t)t * DIM_);
  for (int c = lane; c < 256; c += 32) {
    float4 xv = xr[c];
#pragma unroll
    for (int e = 0; e < NE; ++e) {
      float4 g = gws4[e * 256 + c];
      acc[e] += xv.x * g.x + xv.y * g.y + xv.z * g.z + xv.w * g.w;
    }
  }
#pragma unroll
  for (int e = 0; e < NE; ++e) {
    float v = acc[e];
    for (int off = 16; off >= 1; off >>= 1) v += __shfl_xor(v, off, 32);
    acc[e] = v;
  }
  float m = acc[0];
#pragma unroll
  for (int e = 1; e < NE; ++e) m = fmaxf(m, acc[e]);
  float p[NE], s = 0.f;
#pragma unroll
  for (int e = 0; e < NE; ++e) { p[e] = __expf(acc[e] - m); s += p[e]; }
  float inv = 1.f / s;
  float orig[NE], sc[NE];
#pragma unroll
  for (int e = 0; e < NE; ++e) { orig[e] = p[e] * inv; sc[e] = orig[e] + gb[e]; }
  float gsc[4];
#pragma unroll
  for (int g = 0; g < 4; ++g) {
    float m1 = -1e30f, m2 = -1e30f;
#pragma unroll
    for (int j = 0; j < 4; ++j) {
      float v = sc[g * 4 + j];
      if (v > m1) { m2 = m1; m1 = v; } else if (v > m2) { m2 = v; }
    }
    gsc[g] = m1 + m2;
  }
  int g1 = 0;
#pragma unroll
  for (int g = 1; g < 4; ++g) if (gsc[g] > gsc[g1]) g1 = g;
  int g2 = -1;
#pragma unroll
  for (int g = 0; g < 4; ++g) { if (g == g1) continue; if (g2 < 0 || gsc[g] > gsc[g2]) g2 = g; }
  float msk[NE];
#pragma unroll
  for (int e = 0; e < NE; ++e) {
    int g = e >> 2;
    msk[e] = (g == g1 || g == g2) ? sc[e] : -1e30f;
  }
  int i1 = 0;
#pragma unroll
  for (int e = 1; e < NE; ++e) if (msk[e] > msk[i1]) i1 = e;
  int i2 = -1;
#pragma unroll
  for (int e = 0; e < NE; ++e) { if (e == i1) continue; if (i2 < 0 || msk[e] > msk[i2]) i2 = e; }

  if (lane == 0) {
    idx_out[t * 2 + 0] = i1;
    idx_out[t * 2 + 1] = i2;
    w_out[t * 2 + 0] = orig[i1];
    w_out[t * 2 + 1] = orig[i2];
    atomicAdd(&counts[i1], 1);
    atomicAdd(&counts[i2], 1);
  }
}

// ------------------------- scan & scatter ----------------------------------
__global__ void scan_kernel(const int* __restrict__ counts, int* __restrict__ cursors,
                            int* __restrict__ t2e) {
  if (threadIdx.x == 0 && blockIdx.x == 0) {
    int off = 0;
    for (int e = 0; e < NE; ++e) {
      cursors[e] = off;
      int nt = (counts[e] + 15) >> 4;
      for (int tt = 0; tt < nt; ++tt) t2e[(off >> 4) + tt] = e;
      off += nt << 4;
    }
    for (int tt = off >> 4; tt < MAX_TILES; ++tt) t2e[tt] = -1;
  }
}

__global__ __launch_bounds__(256) void scatter_kernel(
    const int* __restrict__ idxb, const float* __restrict__ wb,
    int* __restrict__ cursors, int* __restrict__ rowmap, float* __restrict__ roww) {
  int t = blockIdx.x * 256 + threadIdx.x;
  if (t >= T_TOK) return;
#pragma unroll
  for (int k = 0; k < 2; ++k) {
    int e = idxb[t * 2 + k];
    int slot = atomicAdd(&cursors[e], 1);
    rowmap[slot] = t;
    roww[slot] = wb[t * 2 + k];
  }
}

// ----------------------- conversion kernels ---------------------------------
__global__ __launch_bounds__(256) void cvt_x_kernel(const float* __restrict__ x,
                                                    unsigned short* __restrict__ xb) {
  size_t i = (size_t)(blockIdx.x * 256 + threadIdx.x) * 4;
  float4 v = *(const float4*)(x + i);
  uint2 o;
  o.x = pack2bf(v.x, v.y);
  o.y = pack2bf(v.z, v.w);
  *(uint2*)(xb + i) = o;
}

// src [K,N] fp32 -> dst [N,K] bf16, per plane (grid.z)
__global__ __launch_bounds__(256) void tcvt_kernel(const float* __restrict__ src,
                                                   unsigned short* __restrict__ dst,
                                                   int K, int N) {
  __shared__ __align__(16) unsigned short tile[64 * LDT];
  size_t plane = (size_t)blockIdx.z * K * N;
  const float* S = src + plane;
  unsigned short* Dd = dst + plane;
  int k0 = blockIdx.y * 64, n0 = blockIdx.x * 64;
  int tid = threadIdx.x;
  int row = tid >> 4, col4 = (tid & 15) << 2;
#pragma unroll
  for (int i = 0; i < 4; ++i) {
    int kk = row + i * 16;
    float4 v = *(const float4*)(S + (size_t)(k0 + kk) * N + n0 + col4);
    tile[(col4 + 0) * LDT + kk] = f2bf(v.x);
    tile[(col4 + 1) * LDT + kk] = f2bf(v.y);
    tile[(col4 + 2) * LDT + kk] = f2bf(v.z);
    tile[(col4 + 3) * LDT + kk] = f2bf(v.w);
  }
  __syncthreads();
  int nn = tid >> 2, kq = (tid & 3) << 4;
  const uint4* p = (const uint4*)&tile[nn * LDT + kq];
  uint4 a = p[0], b = p[1];
  uint4* q = (uint4*)(Dd + (size_t)(n0 + nn) * K + k0 + kq);
  q[0] = a;
  q[1] = b;
}

// -------------------- routed up (W1,W3 -> SiLU -> h) ------------------------
__global__ __launch_bounds__(128) void moe_up_kernel(
    const unsigned short* __restrict__ xb, const unsigned short* __restrict__ Wt1,
    const float* __restrict__ be1, const unsigned short* __restrict__ Wt3,
    const float* __restrict__ be3, const int* __restrict__ t2e,
    const int* __restrict__ rowmap, const unsigned short* __restrict__ zrow,
    unsigned short* __restrict__ hbuf) {
  int rt = blockIdx.y;
  int e = t2e[rt];
  if (e < 0) return;
  __shared__ __align__(16) unsigned short As[2][16 * LDT];
  __shared__ __align__(16) unsigned short B1s[2][64 * LDT];
  __shared__ __align__(16) unsigned short B3s[2][64 * LDT];
  int tid = threadIdx.x, lane = tid & 31, wave = tid >> 5;
  int n0 = blockIdx.x * 64;
  int ar = tid >> 3, ac8 = (tid & 7) << 3;
  int tok = rowmap[rt * 16 + ar];
  const unsigned short* arow = (tok >= 0) ? (xb + (size_t)tok * DIM_) : zrow;
  const unsigned short* W1b = Wt1 + ((size_t)e * INTER_ + n0) * DIM_;
  const unsigned short* W3b = Wt3 + ((size_t)e * INTER_ + n0) * DIM_;

  auto stage = [&](int buf, int k0) {
    cp16(&As[buf][ar * LDT + ac8], arow + k0 + ac8);
    stageB64(B1s[buf], W1b, DIM_, k0, tid);
    stageB64(B3s[buf], W3b, DIM_, k0, tid);
  };
  stage(0, 0);

  v8f acc1 = {0.f, 0.f, 0.f, 0.f, 0.f, 0.f, 0.f, 0.f};
  v8f acc3 = {0.f, 0.f, 0.f, 0.f, 0.f, 0.f, 0.f, 0.f};
  const int NK = DIM_ / KT;  // 16
  for (int i = 0; i < NK; ++i) {
    int cur = i & 1;
    if (i + 1 < NK) {
      stage(1 - cur, (i + 1) * KT);
      waitAsync<9>();
      __builtin_prefetch(W1b + (size_t)(i + 1) * KT, 0, 1);
    } else {
      waitAsync<0>();
    }
    __syncthreads();
#pragma unroll
    for (int kh = 0; kh < 2; ++kh) {
      v16bf a  = fragA(As[cur], lane, kh);
      v16bf b1 = fragB(B1s[cur], wave * 16, lane, kh);
      v16bf b3 = fragB(B3s[cur], wave * 16, lane, kh);
      acc1 = WMMA_BF16(a, b1, acc1);
      acc3 = WMMA_BF16(a, b3, acc3);
    }
    __syncthreads();
  }
  int Nl = lane & 15, Mh = (lane >> 4) * 8;
  int col = n0 + wave * 16 + Nl;
  float b1b = be1[e * INTER_ + col];
  float b3b = be3[e * INTER_ + col];
#pragma unroll
  for (int r = 0; r < 8; ++r) {
    int slotrow = rt * 16 + Mh + r;
    float v1 = acc1[r] + b1b;
    float v3 = acc3[r] + b3b;
    float hv = v1 / (1.f + __expf(-v1)) * v3;
    hbuf[(size_t)slotrow * INTER_ + col] = f2bf(hv);
  }
}

// -------------- routed down (W2, weighted atomic combine) -------------------
__global__ __launch_bounds__(128) void moe_down_kernel(
    const unsigned short* __restrict__ hbuf, const unsigned short* __restrict__ Wt2,
    const float* __restrict__ be2, const int* __restrict__ t2e,
    const int* __restrict__ rowmap, const float* __restrict__ roww,
    float* __restrict__ out) {
  int rt = blockIdx.y;
  int e = t2e[rt];
  if (e < 0) return;
  __shared__ __align__(16) unsigned short As[2][16 * LDT];
  __shared__ __align__(16) unsigned short Bs[2][64 * LDT];
  int tid = threadIdx.x, lane = tid & 31, wave = tid >> 5;
  int n0 = blockIdx.x * 64;
  int ar = tid >> 3, ac8 = (tid & 7) << 3;
  const unsigned short* arow = hbuf + (size_t)(rt * 16 + ar) * INTER_;
  const unsigned short* W2b = Wt2 + ((size_t)e * DIM_ + n0) * INTER_;

  auto stage = [&](int buf, int k0) {
    cp16(&As[buf][ar * LDT + ac8], arow + k0 + ac8);
    stageB64(Bs[buf], W2b, INTER_, k0, tid);
  };
  stage(0, 0);

  v8f acc = {0.f, 0.f, 0.f, 0.f, 0.f, 0.f, 0.f, 0.f};
  const int NK = INTER_ / KT;  // 8
  for (int i = 0; i < NK; ++i) {
    int cur = i & 1;
    if (i + 1 < NK) {
      stage(1 - cur, (i + 1) * KT);
      waitAsync<5>();
      __builtin_prefetch(W2b + (size_t)(i + 1) * KT, 0, 1);
    } else {
      waitAsync<0>();
    }
    __syncthreads();
#pragma unroll
    for (int kh = 0; kh < 2; ++kh) {
      v16bf a = fragA(As[cur], lane, kh);
      v16bf b = fragB(Bs[cur], wave * 16, lane, kh);
      acc = WMMA_BF16(a, b, acc);
    }
    __syncthreads();
  }
  int Nl = lane & 15, Mh = (lane >> 4) * 8;
  int col = n0 + wave * 16 + Nl;
  float bb = be2[e * DIM_ + col];
#pragma unroll
  for (int r = 0; r < 8; ++r) {
    int slot = rt * 16 + Mh + r;
    int tok = rowmap[slot];
    if (tok < 0) continue;
    float wgt = roww[slot];
    unsafeAtomicAdd(&out[(size_t)tok * DIM_ + col], (acc[r] + bb) * wgt);
  }
}

// --------------------------- shared expert ---------------------------------
__global__ __launch_bounds__(128) void shared_up_kernel(
    const unsigned short* __restrict__ xb, const unsigned short* __restrict__ wt1,
    const float* __restrict__ bs1, const unsigned short* __restrict__ wt3,
    const float* __restrict__ bs3, unsigned short* __restrict__ hsbuf) {
  int rt = blockIdx.y;
  __shared__ __align__(16) unsigned short As[2][16 * LDT];
  __shared__ __align__(16) unsigned short B1s[2][64 * LDT];
  __shared__ __align__(16) unsigned short B3s[2][64 * LDT];
  int tid = threadIdx.x, lane = tid & 31, wave = tid >> 5;
  int n0 = blockIdx.x * 64;
  int ar = tid >> 3, ac8 = (tid & 7) << 3;
  const unsigned short* arow = xb + (size_t)(rt * 16 + ar) * DIM_;
  const unsigned short* W1b = wt1 + (size_t)n0 * DIM_;
  const unsigned short* W3b = wt3 + (size_t)n0 * DIM_;

  auto stage = [&](int buf, int k0) {
    cp16(&As[buf][ar * LDT + ac8], arow + k0 + ac8);
    stageB64(B1s[buf], W1b, DIM_, k0, tid);
    stageB64(B3s[buf], W3b, DIM_, k0, tid);
  };
  stage(0, 0);

  v8f acc1 = {0.f, 0.f, 0.f, 0.f, 0.f, 0.f, 0.f, 0.f};
  v8f acc3 = {0.f, 0.f, 0.f, 0.f, 0.f, 0.f, 0.f, 0.f};
  const int NK = DIM_ / KT;
  for (int i = 0; i < NK; ++i) {
    int cur = i & 1;
    if (i + 1 < NK) {
      stage(1 - cur, (i + 1) * KT);
      waitAsync<9>();
    } else {
      waitAsync<0>();
    }
    __syncthreads();
#pragma unroll
    for (int kh = 0; kh < 2; ++kh) {
      v16bf a  = fragA(As[cur], lane, kh);
      v16bf b1 = fragB(B1s[cur], wave * 16, lane, kh);
      v16bf b3 = fragB(B3s[cur], wave * 16, lane, kh);
      acc1 = WMMA_BF16(a, b1, acc1);
      acc3 = WMMA_BF16(a, b3, acc3);
    }
    __syncthreads();
  }
  int Nl = lane & 15, Mh = (lane >> 4) * 8;
  int col = n0 + wave * 16 + Nl;
  float b1b = bs1[col], b3b = bs3[col];
#pragma unroll
  for (int r = 0; r < 8; ++r) {
    int t = rt * 16 + Mh + r;
    float v1 = acc1[r] + b1b;
    float v3 = acc3[r] + b3b;
    float hv = v1 / (1.f + __expf(-v1)) * v3;
    hsbuf[(size_t)t * NSHI + col] = f2bf(hv);
  }
}

__global__ __launch_bounds__(128) void shared_down_kernel(
    const unsigned short* __restrict__ hsbuf, const unsigned short* __restrict__ wt2,
    const float* __restrict__ bs2, float* __restrict__ out) {
  int rt = blockIdx.y;
  __shared__ __align__(16) unsigned short As[2][16 * LDT];
  __shared__ __align__(16) unsigned short Bs[2][64 * LDT];
  int tid = threadIdx.x, lane = tid & 31, wave = tid >> 5;
  int n0 = blockIdx.x * 64;
  int ar = tid >> 3, ac8 = (tid & 7) << 3;
  const unsigned short* arow = hsbuf + (size_t)(rt * 16 + ar) * NSHI;
  const unsigned short* W2b = wt2 + (size_t)n0 * NSHI;

  auto stage = [&](int buf, int k0) {
    cp16(&As[buf][ar * LDT + ac8], arow + k0 + ac8);
    stageB64(Bs[buf], W2b, NSHI, k0, tid);
  };
  stage(0, 0);

  v8f acc = {0.f, 0.f, 0.f, 0.f, 0.f, 0.f, 0.f, 0.f};
  const int NK = NSHI / KT;
  for (int i = 0; i < NK; ++i) {
    int cur = i & 1;
    if (i + 1 < NK) {
      stage(1 - cur, (i + 1) * KT);
      waitAsync<5>();
    } else {
      waitAsync<0>();
    }
    __syncthreads();
#pragma unroll
    for (int kh = 0; kh < 2; ++kh) {
      v16bf a = fragA(As[cur], lane, kh);
      v16bf b = fragB(Bs[cur], wave * 16, lane, kh);
      acc = WMMA_BF16(a, b, acc);
    }
    __syncthreads();
  }
  int Nl = lane & 15, Mh = (lane >> 4) * 8;
  int col = n0 + wave * 16 + Nl;
  float bb = bs2[col];
#pragma unroll
  for (int r = 0; r < 8; ++r) {
    int t = rt * 16 + Mh + r;
    out[(size_t)t * DIM_ + col] = acc[r] + bb;
  }
}

// ------------------------------ launcher -----------------------------------
// workspace layout (bytes), total ~82.2 MB
#define O_IDX    0ull
#define O_W      32768ull
#define O_CNT    65536ull
#define O_CUR    65600ull
#define O_T2E    65664ull
#define O_ROWMAP 67840ull
#define O_ROWW   101632ull
#define O_ZERO   135424ull
#define O_XB     137472ull
#define O_HBUF   8526080ull
#define O_HSBUF  17176832ull
#define O_WT1    25565440ull
#define O_WT3    42342656ull
#define O_WT2    59119872ull
#define O_WS1T   75897088ull
#define O_WS3T   77994240ull
#define O_WS2T   80091392ull

extern "C" void kernel_launch(void* const* d_in, const int* in_sizes, int n_in,
                              void* d_out, int out_size, void* d_ws, size_t ws_size,
                              hipStream_t stream) {
  const float* x      = (const float*)d_in[0];
  const float* gate_w = (const float*)d_in[1];
  const float* gate_b = (const float*)d_in[2];
  const float* We1    = (const float*)d_in[3];
  const float* be1    = (const float*)d_in[4];
  const float* We2    = (const float*)d_in[5];
  const float* be2    = (const float*)d_in[6];
  const float* We3    = (const float*)d_in[7];
  const float* be3    = (const float*)d_in[8];
  const float* ws1    = (const float*)d_in[9];
  const float* bs1    = (const float*)d_in[10];
  const float* ws2    = (const float*)d_in[11];
  const float* bs2    = (const float*)d_in[12];
  const float* ws3    = (const float*)d_in[13];
  const float* bs3    = (const float*)d_in[14];
  float* out = (float*)d_out;

  char* ws = (char*)d_ws;
  int*            idxb    = (int*)(ws + O_IDX);
  float*          wb      = (float*)(ws + O_W);
  int*            counts  = (int*)(ws + O_CNT);
  int*            cursors = (int*)(ws + O_CUR);
  int*            t2e     = (int*)(ws + O_T2E);
  int*            rowmap  = (int*)(ws + O_ROWMAP);
  float*          roww    = (float*)(ws + O_ROWW);
  unsigned short* zrow    = (unsigned short*)(ws + O_ZERO);
  unsigned short* xb      = (unsigned short*)(ws + O_XB);
  unsigned short* hbuf    = (unsigned short*)(ws + O_HBUF);
  unsigned short* hsbuf   = (unsigned short*)(ws + O_HSBUF);
  unsigned short* Wt1     = (unsigned short*)(ws + O_WT1);
  unsigned short* Wt3     = (unsigned short*)(ws + O_WT3);
  unsigned short* Wt2     = (unsigned short*)(ws + O_WT2);
  unsigned short* ws1t    = (unsigned short*)(ws + O_WS1T);
  unsigned short* ws3t    = (unsigned short*)(ws + O_WS3T);
  unsigned short* ws2t    = (unsigned short*)(ws + O_WS2T);

  (void)hipMemsetAsync(counts, 0, 64, stream);
  (void)hipMemsetAsync(rowmap, 0xFF, MAX_TILES * 16 * sizeof(int), stream);
  (void)hipMemsetAsync(zrow, 0, DIM_ * sizeof(unsigned short), stream);

  // one-time per launch: bf16 conversion (+transpose for weights)
  cvt_x_kernel<<<T_TOK * DIM_ / 1024, 256, 0, stream>>>(x, xb);
  tcvt_kernel<<<dim3(INTER_ / 64, DIM_ / 64, NE), 256, 0, stream>>>(We1, Wt1, DIM_, INTER_);
  tcvt_kernel<<<dim3(INTER_ / 64, DIM_ / 64, NE), 256, 0, stream>>>(We3, Wt3, DIM_, INTER_);
  tcvt_kernel<<<dim3(DIM_ / 64, INTER_ / 64, NE), 256, 0, stream>>>(We2, Wt2, INTER_, DIM_);
  tcvt_kernel<<<dim3(NSHI / 64, DIM_ / 64, 1), 256, 0, stream>>>(ws1, ws1t, DIM_, NSHI);
  tcvt_kernel<<<dim3(NSHI / 64, DIM_ / 64, 1), 256, 0, stream>>>(ws3, ws3t, DIM_, NSHI);
  tcvt_kernel<<<dim3(DIM_ / 64, NSHI / 64, 1), 256, 0, stream>>>(ws2, ws2t, NSHI, DIM_);

  // routing
  gate_kernel<<<T_TOK / 8, 256, 0, stream>>>(x, gate_w, gate_b, idxb, wb, counts);
  scan_kernel<<<1, 32, 0, stream>>>(counts, cursors, t2e);
  scatter_kernel<<<T_TOK / 256, 256, 0, stream>>>(idxb, wb, cursors, rowmap, roww);

  // expert GEMMs
  moe_up_kernel<<<dim3(INTER_ / 64, MAX_TILES), 128, 0, stream>>>(
      xb, Wt1, be1, Wt3, be3, t2e, rowmap, zrow, hbuf);
  shared_up_kernel<<<dim3(NSHI / 64, T_TOK / 16), 128, 0, stream>>>(
      xb, ws1t, bs1, ws3t, bs3, hsbuf);
  shared_down_kernel<<<dim3(DIM_ / 64, T_TOK / 16), 128, 0, stream>>>(
      hsbuf, ws2t, bs2, out);
  moe_down_kernel<<<dim3(DIM_ / 64, MAX_TILES), 128, 0, stream>>>(
      hbuf, Wt2, be2, t2e, rowmap, roww, out);
}
